// VGG_Cifar10_68934225101570
// MI455X (gfx1250) — 8-bit matrix operand order repaired and checked by reading, compiled, not run on hardware
//
#include <hip/hip_runtime.h>

typedef __attribute__((ext_vector_type(8))) int v8i;
typedef __attribute__((ext_vector_type(2))) int v2i;

#define EPS 1e-5f

__device__ __forceinline__ signed char fsign(float z) {
    return (signed char)((z > 0.f) - (z < 0.f));
}

__device__ __forceinline__ int aKoff(int v, int half) {
    return ((v >> 2) << 5) + (((v >> 1) & 1) << 4) + (half << 3) + ((v & 1) << 2);
}

__device__ __forceinline__ int bKoff(int v, int lane) {
    return ((v >> 2) << 5) + (((v >> 1) & 1) << 4) + ((lane >> 4) << 3) + ((v & 1) << 2);
}

__device__ __forceinline__ void async_ld_b64(const void* g, void* l) {
#if __has_builtin(__builtin_amdgcn_global_load_async_to_lds_b64)
    __builtin_amdgcn_global_load_async_to_lds_b64(
        (__attribute__((address_space(1))) v2i*)g,
        (__attribute__((address_space(3))) v2i*)l, 0, 0);
#else
    unsigned lo = (unsigned)(__UINTPTR_TYPE__)(__attribute__((address_space(3))) void*)l;
    asm volatile("global_load_async_to_lds_b64 %0, %1, off"
                 :: "v"(lo), "v"(g) : "memory");
#endif
}

template <int N>
__device__ __forceinline__ void wait_async() {
#if __has_builtin(__builtin_amdgcn_s_wait_asynccnt)
    __builtin_amdgcn_s_wait_asynccnt(N);
#else
    asm volatile("s_wait_asynccnt %0" :: "i"(N) : "memory");
#endif
}

__global__ void zero_pad64(unsigned long long* z) {
    if (threadIdx.x < 8) z[threadIdx.x] = 0ull;
}

__global__ void conv1_bin(const float* __restrict__ x, const float* __restrict__ w,
                          const float* __restrict__ bias,
                          const float* __restrict__ g, const float* __restrict__ be,
                          const float* __restrict__ mu, const float* __restrict__ va,
                          signed char* __restrict__ actOut) {
    int idx = blockIdx.x * blockDim.x + threadIdx.x;
    if (idx >= 32 * 32 * 32 * 384) return;
    int o = idx % 384; int r = idx / 384;
    int ww = r % 32; r /= 32;
    int hh = r % 32; int n = r / 32;
    float acc = bias[o];
    for (int c = 0; c < 3; ++c)
        for (int dy = -1; dy <= 1; ++dy) {
            int hs = hh + dy; if (hs < 0 || hs >= 32) continue;
            for (int dx = -1; dx <= 1; ++dx) {
                int wsx = ww + dx; if (wsx < 0 || wsx >= 32) continue;
                float wv = w[((o * 3 + c) * 3 + (dy + 1)) * 3 + (dx + 1)];
                float sw = (wv > 0.f) ? 1.f : ((wv < 0.f) ? -1.f : 0.f);
                acc += x[((n * 3 + c) * 32 + hs) * 32 + wsx] * sw;
            }
        }
    float sc = g[o] * rsqrtf(va[o] + EPS);
    float z = (acc - mu[o]) * sc + be[o];
    z = fminf(1.f, fmaxf(-1.f, z));
    actOut[((size_t)((n * 32 + hh) * 32 + ww)) * 384 + o] = fsign(z);
}

__global__ void pack_conv_w(const float* __restrict__ w, signed char* __restrict__ pB,
                            int Cin, int Cout) {
    int nOb = Cout >> 4;
    int total = ((9 * Cin) >> 6) * nOb * 1024;
    int idx = blockIdx.x * blockDim.x + threadIdx.x;
    if (idx >= total) return;
    int inner = idx & 1023; int blk = idx >> 10;
    int L = inner >> 5;
    int v = (inner >> 2) & 7;
    int b = inner & 3;
    int kb = blk / nOb, ob = blk - kb * nOb;
    int K = (kb << 6) + bKoff(v, L) + b;
    int N = (ob << 4) + (L & 15);
    int t = K / Cin;
    int c = K - t * Cin;
    float wv = w[(size_t)(N * Cin + c) * 9 + t];
    pB[idx] = (signed char)((wv > 0.f) - (wv < 0.f));
}

__global__ void pack_fc_w(const float* __restrict__ w, signed char* __restrict__ pB,
                          int K, int N, int Ntrue, int remap) {
    int nOb = N >> 4;
    int total = (K >> 6) * nOb * 1024;
    int idx = blockIdx.x * blockDim.x + threadIdx.x;
    if (idx >= total) return;
    int inner = idx & 1023; int blk = idx >> 10;
    int L = inner >> 5;
    int v = (inner >> 2) & 7;
    int b = inner & 3;
    int kb = blk / nOb, ob = blk - kb * nOb;
    int kp = (kb << 6) + bKoff(v, L) + b;
    int N_ = (ob << 4) + (L & 15);
    if (N_ >= Ntrue) { pB[idx] = 0; return; }
    int k = remap ? ((kp & 511) * 16 + (kp >> 9)) : kp;
    float wv = w[(size_t)N_ * K + k];
    pB[idx] = (signed char)((wv > 0.f) - (wv < 0.f));
}

template <int POOL>
__global__ void conv_gemm(const signed char* __restrict__ actIn,
                          const signed char* __restrict__ pB,
                          const signed char* __restrict__ zpad,
                          const float* __restrict__ bias,
                          const float* __restrict__ g, const float* __restrict__ be,
                          const float* __restrict__ mu, const float* __restrict__ va,
                          signed char* __restrict__ actOut,
                          int Cin, int Cout, int Hin, int Win, int Ho, int Wo) {
    constexpr int NF = POOL ? 4 : 1;
    __shared__ __align__(16) unsigned char lA[2][NF][16][64];

    const int tid = threadIdx.x;
    const int lane = tid & 31;
    const int wave = tid >> 5;
    const int mt = blockIdx.x;
    const int oblock = blockIdx.y * 4 + wave;
    const int nOb = Cout >> 4;
    const int nK = (9 * Cin) >> 6;
    const int HW = Ho * Wo;
    const int half = lane >> 4;
    const int l15 = lane & 15;

    const int srow = (tid >> 3) & 15;
    const int sseg = tid & 7;

    v8i acc[NF];
#pragma unroll
    for (int f = 0; f < NF; ++f) acc[f] = (v8i){0, 0, 0, 0, 0, 0, 0, 0};

    auto stage = [&](int kk, int buf) {
        int t = (kk << 6) / Cin;
        int c0 = (kk << 6) - t * Cin;
        int dy = t / 3 - 1, dx = t % 3 - 1;
        int m = mt * 16 + srow;
        int n = m / HW; int rem = m - n * HW;
        int oh = rem / Wo, ow = rem - (rem / Wo) * Wo;
#pragma unroll
        for (int f = 0; f < NF; ++f) {
            int h  = POOL ? (oh * 2 + (f >> 1)) : oh;
            int wq = POOL ? (ow * 2 + (f & 1)) : ow;
            int hs = h + dy, wsx = wq + dx;
            const signed char* src = zpad;
            if (hs >= 0 && hs < Hin && wsx >= 0 && wsx < Win)
                src = actIn + (size_t)((n * Hin + hs) * Win + wsx) * Cin + c0 + sseg * 8;
            async_ld_b64(src, &lA[buf][f][srow][sseg * 8]);
        }
    };

    stage(0, 0);
    for (int kk = 0; kk < nK; ++kk) {
        int buf = kk & 1;
        if (kk + 1 < nK) {
            stage(kk + 1, buf ^ 1);
            __builtin_prefetch(pB + ((size_t)((kk + 1) * nOb + oblock) * 32 + lane) * 32, 0, 3);
            wait_async<NF>();
        } else {
            wait_async<0>();
        }
        __syncthreads();

        const int4* bp = (const int4*)(pB + ((size_t)(kk * nOb + oblock) * 32 + lane) * 32);
        int4 q0 = bp[0], q1 = bp[1];
        v8i bf;
        bf[0] = q0.x; bf[1] = q0.y; bf[2] = q0.z; bf[3] = q0.w;
        bf[4] = q1.x; bf[5] = q1.y; bf[6] = q1.z; bf[7] = q1.w;

#pragma unroll
        for (int f = 0; f < NF; ++f) {
            v8i af;
#pragma unroll
            for (int v = 0; v < 8; ++v)
                af[v] = *(const int*)(&lA[buf][f][l15][aKoff(v, half)]);
            acc[f] = __builtin_amdgcn_wmma_i32_16x16x64_iu8(
                true, af, true, bf, acc[f], false, false);
        }
        __syncthreads();
    }

    int o = (oblock << 4) + l15;
    float bi = bias[o];
    float sc = g[o] * rsqrtf(va[o] + EPS);
    float mo = mu[o], bo = be[o];
#pragma unroll
    for (int r = 0; r < 8; ++r) {
        int m = mt * 16 + r + (half << 3);
        int n = m / HW; int rem = m - n * HW;
        int oh = rem / Wo, ow = rem - (rem / Wo) * Wo;
        int vI = acc[0][r];
        if constexpr (POOL) {
            int a1 = acc[1][r], a2 = acc[2][r], a3 = acc[3][r];
            vI = vI > a1 ? vI : a1;
            vI = vI > a2 ? vI : a2;
            vI = vI > a3 ? vI : a3;
        }
        float z = ((float)vI + bi - mo) * sc + bo;
        z = fminf(1.f, fmaxf(-1.f, z));
        actOut[(size_t)((n * Ho + oh) * Wo + ow) * Cout + o] = fsign(z);
    }
}

__global__ void fc_gemm_bn(const signed char* __restrict__ act,
                           const signed char* __restrict__ pB,
                           const float* __restrict__ bias,
                           const float* __restrict__ g, const float* __restrict__ be,
                           const float* __restrict__ mu, const float* __restrict__ va,
                           signed char* __restrict__ actOut, int K, int N) {
    const int lane = threadIdx.x & 31;
    const int mt = blockIdx.x, ob = blockIdx.y;
    const int nOb = N >> 4;
    const int half = lane >> 4, l15 = lane & 15;

    v8i acc = (v8i){0, 0, 0, 0, 0, 0, 0, 0};
    int mrowA = mt * 16 + l15;
    for (int kk = 0; kk < (K >> 6); ++kk) {
        __builtin_prefetch(pB + ((size_t)((kk + 1) * nOb + ob) * 32 + lane) * 32, 0, 3);
        v8i af;
#pragma unroll
        for (int v = 0; v < 8; ++v)
            af[v] = *(const int*)(act + (size_t)mrowA * K + (kk << 6) + aKoff(v, half));
        const int4* bp = (const int4*)(pB + ((size_t)(kk * nOb + ob) * 32 + lane) * 32);
        int4 q0 = bp[0], q1 = bp[1];
        v8i bf;
        bf[0] = q0.x; bf[1] = q0.y; bf[2] = q0.z; bf[3] = q0.w;
        bf[4] = q1.x; bf[5] = q1.y; bf[6] = q1.z; bf[7] = q1.w;
        acc = __builtin_amdgcn_wmma_i32_16x16x64_iu8(true, af, true, bf, acc, false, false);
    }
    int o = (ob << 4) + l15;
    float bi = bias[o];
    float sc = g[o] * rsqrtf(va[o] + EPS);
    float mo = mu[o], bo = be[o];
#pragma unroll
    for (int r = 0; r < 8; ++r) {
        int m = mt * 16 + r + (half << 3);
        float z = ((float)acc[r] + bi - mo) * sc + bo;
        z = fminf(1.f, fmaxf(-1.f, z));
        actOut[(size_t)m * N + o] = fsign(z);
    }
}

__global__ void fc_gemm_logits(const signed char* __restrict__ act,
                               const signed char* __restrict__ pB,
                               const float* __restrict__ lb,
                               const float* __restrict__ m9,
                               const float* __restrict__ v9,
                               float* __restrict__ logits, int K) {
    const int lane = threadIdx.x & 31;
    const int mt = blockIdx.x;
    const int half = lane >> 4, l15 = lane & 15;

    v8i acc = (v8i){0, 0, 0, 0, 0, 0, 0, 0};
    int mrowA = mt * 16 + l15;
    for (int kk = 0; kk < (K >> 6); ++kk) {
        v8i af;
#pragma unroll
        for (int v = 0; v < 8; ++v)
            af[v] = *(const int*)(act + (size_t)mrowA * K + (kk << 6) + aKoff(v, half));
        const int4* bp = (const int4*)(pB + ((size_t)kk * 32 + lane) * 32);
        int4 q0 = bp[0], q1 = bp[1];
        v8i bf;
        bf[0] = q0.x; bf[1] = q0.y; bf[2] = q0.z; bf[3] = q0.w;
        bf[4] = q1.x; bf[5] = q1.y; bf[6] = q1.z; bf[7] = q1.w;
        acc = __builtin_amdgcn_wmma_i32_16x16x64_iu8(true, af, true, bf, acc, false, false);
    }
    if (l15 < 10) {
        float bi = lb[l15];
        float sc = rsqrtf(v9[l15] + EPS);
        float mo = m9[l15];
#pragma unroll
        for (int r = 0; r < 8; ++r) {
            int m = mt * 16 + r + (half << 3);
            logits[m * 10 + l15] = ((float)acc[r] + bi - mo) * sc;
        }
    }
}

__global__ void logsoftmax10(const float* __restrict__ logits, float* __restrict__ out) {
    int r = threadIdx.x;
    if (r >= 32) return;
    float v[10], mx = -1e30f;
    for (int j = 0; j < 10; ++j) { v[j] = logits[r * 10 + j]; mx = fmaxf(mx, v[j]); }
    float s = 0.f;
    for (int j = 0; j < 10; ++j) s += __expf(v[j] - mx);
    float l = __logf(s) + mx;
    for (int j = 0; j < 10; ++j) out[r * 10 + j] = v[j] - l;
}

extern "C" void kernel_launch(void* const* d_in, const int* in_sizes, int n_in,
                              void* d_out, int out_size, void* d_ws, size_t ws_size,
                              hipStream_t stream) {
    (void)in_sizes; (void)n_in; (void)out_size; (void)ws_size;

    const float* x = (const float*)d_in[0];
    const float* cw[6]; const float* cb[6];
    for (int i = 0; i < 6; ++i) { cw[i] = (const float*)d_in[1 + 2 * i]; cb[i] = (const float*)d_in[2 + 2 * i]; }
    const float *G[8], *B[8], *M[8], *V[8];
    for (int i = 0; i < 8; ++i) {
        G[i] = (const float*)d_in[13 + 4 * i];
        B[i] = (const float*)d_in[14 + 4 * i];
        M[i] = (const float*)d_in[15 + 4 * i];
        V[i] = (const float*)d_in[16 + 4 * i];
    }
    const float* m9 = (const float*)d_in[45];
    const float* v9 = (const float*)d_in[46];
    const float* lw[3]; const float* lb[3];
    for (int i = 0; i < 3; ++i) { lw[i] = (const float*)d_in[47 + 2 * i]; lb[i] = (const float*)d_in[48 + 2 * i]; }

    char* ws = (char*)d_ws;
    signed char* actA   = (signed char*)(ws);
    signed char* actB   = (signed char*)(ws + (13ull << 20));
    signed char* wpk    = (signed char*)(ws + (17ull << 20));
    float*       logits = (float*)(ws + (28ull << 20));
    signed char* zpad   = (signed char*)(ws + (28ull << 20) + 16384);

    zero_pad64<<<1, 64, 0, stream>>>((unsigned long long*)zpad);

    {
        int total = 32 * 32 * 32 * 384;
        conv1_bin<<<(total + 255) / 256, 256, 0, stream>>>(
            x, cw[0], cb[0], G[0], B[0], M[0], V[0], actA);
    }

    struct CL { int cwi, Cin, Cout, Hin, pool, bn; };
    const CL Ls[5] = {
        {1, 384,  384, 32, 1, 1},
        {2, 384,  768, 16, 0, 2},
        {3, 768,  768, 16, 1, 3},
        {4, 768, 1536,  8, 0, 4},
        {5, 1536, 512,  8, 1, 5},
    };

    signed char* cur = actA;
    signed char* nxt = actB;
    for (int li = 0; li < 5; ++li) {
        const CL& L = Ls[li];
        int nK = (9 * L.Cin) >> 6, nOb = L.Cout >> 4;
        int totalPk = nK * nOb * 1024;
        pack_conv_w<<<(totalPk + 255) / 256, 256, 0, stream>>>(cw[L.cwi], wpk, L.Cin, L.Cout);

        int Ho = L.pool ? L.Hin / 2 : L.Hin;
        int Mt = 32 * Ho * Ho / 16;
        dim3 grid(Mt, L.Cout / 64);
        if (L.pool)
            conv_gemm<1><<<grid, 128, 0, stream>>>(cur, wpk, zpad, cb[L.cwi],
                G[L.bn], B[L.bn], M[L.bn], V[L.bn], nxt,
                L.Cin, L.Cout, L.Hin, L.Hin, Ho, Ho);
        else
            conv_gemm<0><<<grid, 128, 0, stream>>>(cur, wpk, zpad, cb[L.cwi],
                G[L.bn], B[L.bn], M[L.bn], V[L.bn], nxt,
                L.Cin, L.Cout, L.Hin, L.Hin, Ho, Ho);
        signed char* t = cur; cur = nxt; nxt = t;
    }

    {
        int totalPk = (8192 / 64) * (1024 / 16) * 1024;
        pack_fc_w<<<(totalPk + 255) / 256, 256, 0, stream>>>(lw[0], wpk, 8192, 1024, 1024, 1);
        fc_gemm_bn<<<dim3(2, 64), 32, 0, stream>>>(cur, wpk, lb[0],
            G[6], B[6], M[6], V[6], nxt, 8192, 1024);
        signed char* t = cur; cur = nxt; nxt = t;
    }
    {
        int totalPk = (1024 / 64) * (1024 / 16) * 1024;
        pack_fc_w<<<(totalPk + 255) / 256, 256, 0, stream>>>(lw[1], wpk, 1024, 1024, 1024, 0);
        fc_gemm_bn<<<dim3(2, 64), 32, 0, stream>>>(cur, wpk, lb[1],
            G[7], B[7], M[7], V[7], nxt, 1024, 1024);
        signed char* t = cur; cur = nxt; nxt = t;
    }
    {
        int totalPk = (1024 / 64) * 1 * 1024;
        pack_fc_w<<<(totalPk + 255) / 256, 256, 0, stream>>>(lw[2], wpk, 1024, 16, 10, 0);
        fc_gemm_logits<<<dim3(2, 1), 32, 0, stream>>>(cur, wpk, lb[2], m9, v9, logits, 1024);
    }

    logsoftmax10<<<1, 32, 0, stream>>>(logits, (float*)d_out);
}
